// LstmRankPooling_69827578298744
// MI455X (gfx1250) — matrix operand order repaired and checked by reading, not compiled, not run on hardware
//
#include <hip/hip_runtime.h>
#include <hip/hip_bf16.h>

// ---------------------------------------------------------------------------
// LSTM rank pooling for MI455X (gfx1250, wave32, WMMA bf16 16x16x32).
//   B=64, D=768, T=512, 4D=3072
// Phases:
//   P0  prep: W^T -> bf16, bias, x -> bf16 [T][B][D], init h/c/SS
//   P1  512x lstm_step: gates = x_t@Wih^T + h@Whh^T via v_wmma_f32_16x16x32_bf16
//       (fully-unrolled ping-pong pipelined K loops, 4 gate tiles per A fragment)
//   P2  Gram K[b] = Xn^T Xn via WMMA (16x32 strip / wave, pipelined, unrolled),
//       l2-norm folded into epilogue; K (64 MB) stays in the 192 MB L2
//   P3  power iteration (20) -> eta  (1 WG / batch, v in LDS, K rows from L2)
//   P4  ISTA (200)           -> beta (1 WG / batch, beta in LDS)
//   P5  w = X beta (norms folded into beta) + final l2 norm
// ---------------------------------------------------------------------------

typedef __attribute__((ext_vector_type(16))) __bf16 bf16x16;
typedef __attribute__((ext_vector_type(8)))  float  v8f;

union AFrag { bf16x16 v; __bf16 e[16]; };

#define B_   64
#define D_   768
#define T_   512
#define G4   3072   /* 4*D */

__device__ __forceinline__ float sigm(float x) { return 1.0f / (1.0f + __expf(-x)); }

// 16x32 bf16 A-matrix VGPR layout (CDNA5 ISA):
// lanes 0-15:  VGPR0-3 K=0..7,  VGPR4-7 K=16..23
// lanes 16-31: VGPR0-3 K=8..15, VGPR4-7 K=24..31   (pairs packed per VGPR)
__device__ __forceinline__ int kpair(int v, int half) {
  return (v < 4) ? (2 * v + 8 * half) : (16 + 2 * (v - 4) + 8 * half);
}

__device__ __forceinline__ void load_afrag(AFrag& a, const __bf16* __restrict__ base,
                                           int k0, int half) {
#pragma unroll
  for (int v = 0; v < 8; ++v) {
    const int kp = k0 + kpair(v, half);
    a.e[2 * v]     = base[kp];
    a.e[2 * v + 1] = base[kp + 1];
  }
}

// ---------------------------------------------------------------------------
// P0: prep kernels
// ---------------------------------------------------------------------------
__global__ void prep_weights_kernel(const float* __restrict__ wih,
                                    const float* __restrict__ whh,
                                    const float* __restrict__ bih,
                                    const float* __restrict__ bhh,
                                    __bf16* __restrict__ twih,
                                    __bf16* __restrict__ twhh,
                                    float* __restrict__ bias) {
  const int id = blockIdx.x * blockDim.x + threadIdx.x;   // over [4D][D]
  if (id < G4 * D_) {
    const int n = id / D_, k = id % D_;
    twih[(size_t)n * D_ + k] = (__bf16)wih[id];           // W : [4D][D]
    twhh[(size_t)n * D_ + k] = (__bf16)whh[id];
  }
  if (id < G4) bias[id] = bih[id] + bhh[id];
}

__global__ void prep_inputs_kernel(const float* __restrict__ in,   // [B][D][T]
                                   __bf16* __restrict__ xin) {     // [T][B][D]
  const int id = blockIdx.x * blockDim.x + threadIdx.x;
  if (id >= B_ * D_ * T_) return;
  const int t = id & (T_ - 1);
  const int d = (id >> 9) % D_;
  const int b = id / (D_ * T_);
  xin[((size_t)t * B_ + b) * D_ + d] = (__bf16)in[id];
}

__global__ void prep_state_kernel(const float* __restrict__ h0,
                                  const float* __restrict__ c0,
                                  __bf16* __restrict__ hb,
                                  float* __restrict__ cst,
                                  float* __restrict__ ss) {
  const int id = blockIdx.x * blockDim.x + threadIdx.x;
  if (id < B_ * D_) { hb[id] = (__bf16)h0[id]; cst[id] = c0[id]; }
  if (id < B_ * T_) ss[id] = 0.0f;
}

// ---------------------------------------------------------------------------
// Fully-unrolled ping-pong pipelined GEMM phase for the LSTM step:
//   acc[g] += A(16x768 from abase) x Wt columns [g*768+d0 .. +16)   (24 K-steps)
// Full unroll -> each stage is fresh SSA, loads land directly in WMMA operands.
// ---------------------------------------------------------------------------
__device__ __forceinline__ void gemm_phase(v8f (&acc)[4],
                                           const __bf16* __restrict__ abase,
                                           const __bf16* __restrict__ wb,
                                           int d0, int lane, int half) {
  AFrag a0, a1;
  bf16x16 b0[4], b1[4];
  load_afrag(a0, abase, 0, half);
  load_afrag(a1, abase, 32, half);
#pragma unroll
  for (int g = 0; g < 4; ++g) {
    AFrag t0, t1;
    const __bf16* wr = wb + (size_t)(g * D_ + d0 + (lane & 15)) * D_;
    load_afrag(t0, wr, 0, half);
    load_afrag(t1, wr, 32, half);
    b0[g] = t0.v;
    b1[g] = t1.v;
  }
#pragma unroll
  for (int kk = 0; kk + 2 < 24; kk += 2) {
    // stage 0: compute k-step kk, refill with k-step kk+2
#pragma unroll
    for (int g = 0; g < 4; ++g)
      acc[g] = __builtin_amdgcn_wmma_f32_16x16x32_bf16(
          false, a0.v, false, b0[g], (short)0, acc[g], false, false);
    load_afrag(a0, abase, (kk + 2) * 32, half);
#pragma unroll
    for (int g = 0; g < 4; ++g) {
      AFrag t;
      load_afrag(t, wb + (size_t)(g * D_ + d0 + (lane & 15)) * D_, (kk + 2) * 32, half);
      b0[g] = t.v;
    }
    // stage 1: compute k-step kk+1, refill with k-step kk+3
#pragma unroll
    for (int g = 0; g < 4; ++g)
      acc[g] = __builtin_amdgcn_wmma_f32_16x16x32_bf16(
          false, a1.v, false, b1[g], (short)0, acc[g], false, false);
    load_afrag(a1, abase, (kk + 3) * 32, half);
#pragma unroll
    for (int g = 0; g < 4; ++g) {
      AFrag t;
      load_afrag(t, wb + (size_t)(g * D_ + d0 + (lane & 15)) * D_, (kk + 3) * 32, half);
      b1[g] = t.v;
    }
  }
  // tail: k-steps 22, 23
#pragma unroll
  for (int g = 0; g < 4; ++g)
    acc[g] = __builtin_amdgcn_wmma_f32_16x16x32_bf16(
        false, a0.v, false, b0[g], (short)0, acc[g], false, false);
#pragma unroll
  for (int g = 0; g < 4; ++g)
    acc[g] = __builtin_amdgcn_wmma_f32_16x16x32_bf16(
        false, a1.v, false, b1[g], (short)0, acc[g], false, false);
}

// ---------------------------------------------------------------------------
// P1: one LSTM step. grid = 24 blocks * 8 waves = 192 waves = 4 Mtiles x 48 Dtiles
// Each wave: 16x16 tile of (batch x feature), all 4 gates.
// ---------------------------------------------------------------------------
__global__ __launch_bounds__(256) void lstm_step_kernel(
    int t,
    const __bf16* __restrict__ xin,   // [T][B][D]
    const __bf16* __restrict__ hin,   // [B][D]
    __bf16* __restrict__ hout,        // [B][D]
    const __bf16* __restrict__ wih,   // [4D][D]
    const __bf16* __restrict__ whh,   // [4D][D]
    const float*  __restrict__ bias,  // [4D]
    float*  __restrict__ cst,         // [B][D]
    __bf16* __restrict__ xb,          // [B][T][D]
    __bf16* __restrict__ xbt,         // [B][D][T]
    float*  __restrict__ ss) {        // [B][T]
  const int lane = threadIdx.x & 31;
  const int half = lane >> 4;
  const int l15  = lane & 15;
  const int wid  = blockIdx.x * 8 + (threadIdx.x >> 5);   // 0..191
  const int m0   = (wid / 48) * 16;                       // batch tile base
  const int d0   = (wid % 48) * 16;                       // feature tile base

  v8f acc[4];
#pragma unroll
  for (int g = 0; g < 4; ++g) {
    const float bv = bias[g * D_ + d0 + l15];             // N depends only on lane
#pragma unroll
    for (int i = 0; i < 8; ++i) acc[g][i] = bv;
  }

  const __bf16* abase1 = xin + ((size_t)t * B_ + m0 + l15) * D_;
  const __bf16* abase2 = hin + (size_t)(m0 + l15) * D_;

  gemm_phase(acc, abase1, wih, d0, lane, half);   // x_t @ W_ih^T
  gemm_phase(acc, abase2, whh, d0, lane, half);   // h_{t-1} @ W_hh^T

  // epilogue: C layout -> VGPR r: lanes0-15 M=r, lanes16-31 M=r+8; N = lane&15
#pragma unroll
  for (int r = 0; r < 8; ++r) {
    const int bb = m0 + r + 8 * half;
    const int d  = d0 + l15;
    const float gi = sigm(acc[0][r]);
    const float gf = sigm(acc[1][r]);
    const float gg = tanhf(acc[2][r]);
    const float go = sigm(acc[3][r]);
    const int ci = bb * D_ + d;
    const float c = gf * cst[ci] + gi * gg;
    cst[ci] = c;
    const float h = go * tanhf(c);
    hout[ci] = (__bf16)h;
    const float x = tanhf(h);
    xb [((size_t)bb * T_ + t) * D_ + d] = (__bf16)x;
    xbt[((size_t)bb * D_ + d) * T_ + t] = (__bf16)x;
    float s2 = x * x;                    // reduce over 16-lane group (same b,t)
    s2 += __shfl_xor(s2, 1, 16);
    s2 += __shfl_xor(s2, 2, 16);
    s2 += __shfl_xor(s2, 4, 16);
    s2 += __shfl_xor(s2, 8, 16);
    if (l15 == 0) atomicAdd(&ss[bb * T_ + t], s2);
  }
}

// ---------------------------------------------------------------------------
// P2: Gram K[b][t][s] = sum_d Xn[b][d][t] Xn[b][d][s]; norm folded in epilogue.
// Each wave computes a 16(t) x 32(s) strip: A fragment reused for 2 WMMAs.
// grid = 4096 blocks * 8 waves = 64 batches * 32 t-tiles * 16 s-pairs.
// ---------------------------------------------------------------------------
__global__ __launch_bounds__(256) void gram_kernel(
    const __bf16* __restrict__ xb,    // [B][T][D]
    const __bf16* __restrict__ xbt,   // [B][D][T]
    const float*  __restrict__ ss,    // [B][T]
    float* __restrict__ km) {         // [B][T][T]
  const int lane = threadIdx.x & 31;
  const int half = lane >> 4;
  const int l15  = lane & 15;
  const int wid  = blockIdx.x * 8 + (threadIdx.x >> 5);   // 0..32767
  const int b  = wid >> 9;           // 512 waves per batch
  const int tt = (wid >> 4) & 31;    // t tile
  const int sp = wid & 15;           // s pair
  const int t0 = tt * 16, s0 = sp * 32;

  v8f acc0, acc1;
#pragma unroll
  for (int i = 0; i < 8; ++i) { acc0[i] = 0.0f; acc1[i] = 0.0f; }

  const __bf16* abase = xb  + ((size_t)b * T_ + t0 + l15) * D_;
  const __bf16* bbase = xb  + ((size_t)b * T_ + s0 + l15) * D_;

  AFrag a0, a1;
  bf16x16 p0, q0, p1, q1;              // (s0, s0+16) B fragments, 2 stages
  load_afrag(a0, abase, 0, half);
  load_afrag(a1, abase, 32, half);
  { AFrag tb;
    load_afrag(tb, bbase, 0, half);            p0 = tb.v;
    load_afrag(tb, bbase + 16 * D_, 0, half);  q0 = tb.v;
    load_afrag(tb, bbase, 32, half);           p1 = tb.v;
    load_afrag(tb, bbase + 16 * D_, 32, half); q1 = tb.v; }

#pragma unroll
  for (int kk = 0; kk + 2 < 24; kk += 2) {
    acc0 = __builtin_amdgcn_wmma_f32_16x16x32_bf16(
        false, a0.v, false, p0, (short)0, acc0, false, false);
    acc1 = __builtin_amdgcn_wmma_f32_16x16x32_bf16(
        false, a0.v, false, q0, (short)0, acc1, false, false);
    load_afrag(a0, abase, (kk + 2) * 32, half);
    { AFrag tb;
      load_afrag(tb, bbase, (kk + 2) * 32, half);           p0 = tb.v;
      load_afrag(tb, bbase + 16 * D_, (kk + 2) * 32, half); q0 = tb.v; }
    acc0 = __builtin_amdgcn_wmma_f32_16x16x32_bf16(
        false, a1.v, false, p1, (short)0, acc0, false, false);
    acc1 = __builtin_amdgcn_wmma_f32_16x16x32_bf16(
        false, a1.v, false, q1, (short)0, acc1, false, false);
    load_afrag(a1, abase, (kk + 3) * 32, half);
    { AFrag tb;
      load_afrag(tb, bbase, (kk + 3) * 32, half);           p1 = tb.v;
      load_afrag(tb, bbase + 16 * D_, (kk + 3) * 32, half); q1 = tb.v; }
  }
  acc0 = __builtin_amdgcn_wmma_f32_16x16x32_bf16(
      false, a0.v, false, p0, (short)0, acc0, false, false);
  acc1 = __builtin_amdgcn_wmma_f32_16x16x32_bf16(
      false, a0.v, false, q0, (short)0, acc1, false, false);
  acc0 = __builtin_amdgcn_wmma_f32_16x16x32_bf16(
      false, a1.v, false, p1, (short)0, acc0, false, false);
  acc1 = __builtin_amdgcn_wmma_f32_16x16x32_bf16(
      false, a1.v, false, q1, (short)0, acc1, false, false);

  const float rns0 = rsqrtf(ss[b * T_ + s0 + l15]      + 1e-12f);
  const float rns1 = rsqrtf(ss[b * T_ + s0 + 16 + l15] + 1e-12f);
#pragma unroll
  for (int r = 0; r < 8; ++r) {
    const int tr = t0 + r + 8 * half;
    const float rnt = rsqrtf(ss[b * T_ + tr] + 1e-12f);
    float* row = km + ((size_t)b * T_ + tr) * T_;
    row[s0 + l15]      = acc0[r] * rnt * rns0;
    row[s0 + 16 + l15] = acc1[r] * rnt * rns1;
  }
}

// ---------------------------------------------------------------------------
// P3: power iteration -> eta.  One workgroup (512 threads) per batch.
// ---------------------------------------------------------------------------
__global__ __launch_bounds__(512) void power_kernel(const float* __restrict__ km,
                                                    float* __restrict__ eta) {
  __shared__ __align__(16) float sv[T_];
  __shared__ __align__(16) float sw[T_];
  __shared__ __align__(16) float sr[T_];
  const int b = blockIdx.x, t = threadIdx.x;
  const float4* krow = reinterpret_cast<const float4*>(km + ((size_t)b * T_ + t) * T_);
  sv[t] = 1.0f;
  __syncthreads();
  for (int it = 0; it < 20; ++it) {
    float s = 0.0f;
    const float4* vv = reinterpret_cast<const float4*>(sv);
    for (int j = 0; j < T_ / 4; ++j) {
      const float4 k4 = krow[j], v4 = vv[j];
      s = fmaf(k4.x, v4.x, s); s = fmaf(k4.y, v4.y, s);
      s = fmaf(k4.z, v4.z, s); s = fmaf(k4.w, v4.w, s);
    }
    sw[t] = s; sr[t] = s * s;
    __syncthreads();
    for (int off = 256; off > 0; off >>= 1) {
      if (t < off) sr[t] += sr[t + off];
      __syncthreads();
    }
    const float nrm = sqrtf(sr[0]);
    __syncthreads();
    sv[t] = sw[t] / (nrm + 1e-12f);     // reference: v / (||v|| + 1e-12)
    __syncthreads();
  }
  // Rayleigh quotient L = v^T K v
  float s = 0.0f;
  const float4* vv = reinterpret_cast<const float4*>(sv);
  for (int j = 0; j < T_ / 4; ++j) {
    const float4 k4 = krow[j], v4 = vv[j];
    s = fmaf(k4.x, v4.x, s); s = fmaf(k4.y, v4.y, s);
    s = fmaf(k4.z, v4.z, s); s = fmaf(k4.w, v4.w, s);
  }
  sr[t] = sv[t] * s;
  __syncthreads();
  for (int off = 256; off > 0; off >>= 1) {
    if (t < off) sr[t] += sr[t + off];
    __syncthreads();
  }
  if (t == 0) eta[b] = 1.0f / (1.01f * sr[0] + 1e-6f);
}

// ---------------------------------------------------------------------------
// P4: ISTA (200 iters) -> beta; 1/norm folded into output beta.
// ---------------------------------------------------------------------------
__global__ __launch_bounds__(512) void ista_kernel(const float* __restrict__ km,
                                                   const float* __restrict__ eta,
                                                   const float* __restrict__ ss,
                                                   float* __restrict__ bs) {
  __shared__ __align__(16) float sb[T_];
  const int b = blockIdx.x, t = threadIdx.x;
  const float4* krow = reinterpret_cast<const float4*>(km + ((size_t)b * T_ + t) * T_);
  const float et = eta[b];
  sb[t] = 0.0f;
  __syncthreads();
  const float y = (float)(t + 1);
  for (int it = 0; it < 200; ++it) {
    float s = 0.0f;
    const float4* bb = reinterpret_cast<const float4*>(sb);
    for (int j = 0; j < T_ / 4; ++j) {
      const float4 k4 = krow[j], b4 = bb[j];
      s = fmaf(k4.x, b4.x, s); s = fmaf(k4.y, b4.y, s);
      s = fmaf(k4.z, b4.z, s); s = fmaf(k4.w, b4.w, s);
    }
    const float z = sb[t] - et * (s - y);           // gradient step
    float az = fmaxf(fabsf(z) - et * 0.1f, 0.0f);   // soft threshold (eps=0.1)
    float nb = (z >= 0.0f) ? az : -az;
    nb = fminf(fmaxf(nb, -1000.0f), 1000.0f);       // box |beta| <= C
    __syncthreads();
    sb[t] = nb;
    __syncthreads();
  }
  bs[b * T_ + t] = sb[t] * rsqrtf(ss[b * T_ + t] + 1e-12f);
}

// ---------------------------------------------------------------------------
// P5: w[b][d] = sum_t X[b][t][d] * beta'[t];  out = w / sqrt(sum w^2 + 1e-12)
// ---------------------------------------------------------------------------
__global__ __launch_bounds__(768) void final_kernel(const __bf16* __restrict__ xb,
                                                    const float* __restrict__ bs,
                                                    float* __restrict__ out) {
  __shared__ float sbeta[T_];
  __shared__ float sred[D_];
  const int b = blockIdx.x, d = threadIdx.x;
  if (d < T_) sbeta[d] = bs[b * T_ + d];
  __syncthreads();
  float acc = 0.0f;
  const __bf16* xp = xb + (size_t)b * T_ * D_ + d;
  for (int t = 0; t < T_; ++t) acc = fmaf((float)xp[(size_t)t * D_], sbeta[t], acc);
  sred[d] = acc * acc;
  __syncthreads();
  if (d < 256) sred[d] += sred[d + 512];            // fold 768 -> 512
  __syncthreads();
  for (int off = 256; off > 0; off >>= 1) {
    if (d < off) sred[d] += sred[d + off];
    __syncthreads();
  }
  out[b * D_ + d] = acc * rsqrtf(sred[0] + 1e-12f);
}

// ---------------------------------------------------------------------------
// Host side
// ---------------------------------------------------------------------------
extern "C" void kernel_launch(void* const* d_in, const int* in_sizes, int n_in,
                              void* d_out, int out_size, void* d_ws, size_t ws_size,
                              hipStream_t stream) {
  const float* inputs = (const float*)d_in[0];   // [B][D][T]
  const float* h0     = (const float*)d_in[1];   // [1][B][D]
  const float* c0     = (const float*)d_in[2];   // [1][B][D]
  const float* W_ih   = (const float*)d_in[3];   // [4D][D]
  const float* W_hh   = (const float*)d_in[4];   // [4D][D]
  const float* b_ih   = (const float*)d_in[5];   // [4D]
  const float* b_hh   = (const float*)d_in[6];   // [4D]
  float* out = (float*)d_out;                    // [B][D]
  (void)in_sizes; (void)n_in; (void)out_size; (void)ws_size;

  char* ws = (char*)d_ws;
  size_t off = 0;
  auto take = [&](size_t bytes) -> char* {
    char* p = ws + off;
    off += (bytes + 255) & ~(size_t)255;
    return p;
  };
  __bf16* TWIH = (__bf16*)take((size_t)D_ * G4 * 2);        //  4.7 MB
  __bf16* TWHH = (__bf16*)take((size_t)D_ * G4 * 2);        //  4.7 MB
  float*  BIAS = (float*) take((size_t)G4 * 4);             //   12 KB
  __bf16* XIN  = (__bf16*)take((size_t)T_ * B_ * D_ * 2);   // 50.3 MB
  __bf16* HB0  = (__bf16*)take((size_t)B_ * D_ * 2);
  __bf16* HB1  = (__bf16*)take((size_t)B_ * D_ * 2);
  float*  CST  = (float*) take((size_t)B_ * D_ * 4);
  __bf16* XB   = (__bf16*)take((size_t)B_ * T_ * D_ * 2);   // 50.3 MB
  __bf16* XBT  = (__bf16*)take((size_t)B_ * D_ * T_ * 2);   // 50.3 MB
  float*  SS   = (float*) take((size_t)B_ * T_ * 4);
  float*  KM   = (float*) take((size_t)B_ * T_ * T_ * 4);   // 67.1 MB (L2-resident)
  float*  ETA  = (float*) take((size_t)B_ * 4);
  float*  BS   = (float*) take((size_t)B_ * T_ * 4);

  prep_weights_kernel<<<(G4 * D_ + 255) / 256, 256, 0, stream>>>(
      W_ih, W_hh, b_ih, b_hh, TWIH, TWHH, BIAS);
  prep_inputs_kernel<<<(B_ * D_ * T_ + 255) / 256, 256, 0, stream>>>(inputs, XIN);
  prep_state_kernel<<<(B_ * D_ + 255) / 256, 256, 0, stream>>>(h0, c0, HB0, CST, SS);

  for (int t = 0; t < T_; ++t) {
    const __bf16* hin = (t & 1) ? HB1 : HB0;
    __bf16* hout      = (t & 1) ? HB0 : HB1;
    lstm_step_kernel<<<24, 256, 0, stream>>>(
        t, XIN, hin, hout, TWIH, TWHH, BIAS, CST, XB, XBT, SS);
  }

  gram_kernel<<<(B_ * 32 * 16) / 8, 256, 0, stream>>>(XB, XBT, SS, KM);
  power_kernel<<<B_, T_, 0, stream>>>(KM, ETA);
  ista_kernel<<<B_, T_, 0, stream>>>(KM, ETA, SS, BS);
  final_kernel<<<B_, D_, 0, stream>>>(XB, BS, out);
}
